// MultiViewEntropyBottleneck_74045236183781
// MI455X (gfx1250) — matrix operand order repaired and checked by reading, not compiled, not run on hardware
//
#include <hip/hip_runtime.h>
#include <hip/hip_bf16.h>
#include <math.h>

// ---------------------------------------------------------------------------
// MultiViewEntropyBottleneck forward for MI455X (gfx1250, wave32, WMMA).
// B=1024 S=64 P=768 H=1024 D=256.
//  - All GEMMs: v_wmma_f32_16x16x32_bf16, bf16 operands in HBM (half traffic),
//    f32 accumulation, optional bf16 store-through for LN-bound activations.
//  - Double-buffered LDS pipeline; A tiles staged via
//    GLOBAL_LOAD_ASYNC_TO_LDS_B128 (+ s_wait_asynccnt) when the builtin exists.
//  - Score MLPs only on the 1024 last-token rows (mask all ones => last=S-1).
//  - Median bandwidth via device-side bisection (graph-capturable).
// ---------------------------------------------------------------------------

#define BDIM   1024
#define SDIM   64
#define PDIM   768
#define HDIM   1024
#define DDIM   256
#define MROWS  (BDIM * SDIM)        // 65536
#define BSD    ((size_t)MROWS * DDIM)

typedef __attribute__((ext_vector_type(16))) __bf16 v16bf;
typedef __attribute__((ext_vector_type(8)))  float  v8f;

#if defined(__has_builtin)
#if __has_builtin(__builtin_amdgcn_global_load_async_to_lds_b128) && \
    __has_builtin(__builtin_amdgcn_s_wait_asynccnt)
#define USE_ASYNC_LDS 1
#endif
#endif
#ifndef USE_ASYNC_LDS
#define USE_ASYNC_LDS 0
#endif

typedef int v4i_async __attribute__((vector_size(16)));
typedef __attribute__((address_space(1))) v4i_async* gptr_v4i;
typedef __attribute__((address_space(3))) v4i_async* lptr_v4i;

struct Scal {
    float klsum[2];
    float alignsum;
    float lo, hi, cnt;
    float inv_2bw2, inv_bw2, inv_bw4;
};

__device__ __forceinline__ unsigned short f2bf(float f) {
    unsigned u = __float_as_uint(f);
    unsigned r = u + 0x7FFFu + ((u >> 16) & 1u);   // round-to-nearest-even
    return (unsigned short)(r >> 16);
}
__device__ __forceinline__ float bf2f(unsigned short h) {
    return __uint_as_float(((unsigned)h) << 16);
}

__device__ __forceinline__ float block_reduce_sum(float v) {
    __shared__ float buf[256];
    int t = threadIdx.x;
    buf[t] = v; __syncthreads();
    for (int s = 128; s > 0; s >>= 1) {
        if (t < s) buf[t] += buf[t + s];
        __syncthreads();
    }
    float r = buf[0]; __syncthreads();
    return r;
}

// ---------------------------------------------------------------------------
// GEMM: C(MxN) = A(MxK,bf16) @ B(KxN,bf16) + bias, f32 accum.
// 256 thr / 8 waves; block tile 128x128; wave tile 32x64; K-step 32.
// All call sites have M%128==0, N%128==0, K%32==0 (no edge guards needed).
// ---------------------------------------------------------------------------
#define LSTR 48   // bf16 units per LDS row (96B: 16B-aligned, staggered banks)

union BF16x16 { v16bf v; uint4 q[2]; };
union F8      { v8f   v; float f[8]; };

__device__ __forceinline__ void copy16_g2l(const unsigned short* g,
                                           unsigned short* l) {
#if USE_ASYNC_LDS
    __builtin_amdgcn_global_load_async_to_lds_b128(
        (gptr_v4i)(void*)g, (lptr_v4i)(void*)l, 0, 0);
#else
    *(uint4*)l = *(const uint4*)g;
#endif
}

__device__ __forceinline__ v16bf load_a_frag(const unsigned short* As,
                                             int rowBase, int lane) {
    // A 16x32 bf16: lanes 0-15 (M=lane): K 0..7,16..23 ; lanes 16-31: K 8..15,24..31
    int m  = lane & 15;
    int kh = lane >> 4;
    const char* p = (const char*)&As[(rowBase + m) * LSTR] + kh * 16;
    BF16x16 f;
    f.q[0] = *(const uint4*)(p);
    f.q[1] = *(const uint4*)(p + 32);
    return f.v;
}

__device__ __forceinline__ v16bf load_b_frag(const unsigned short* Bs,
                                             int colBase, int lane) {
    // B 32x16 bf16: lanes 0-15 (N=lane): K 0..7,16..23 ; lanes 16-31: K 8..15,24..31.
    // Bs is the tile transposed: Bs[n][k], contiguous in k.
    int n   = lane & 15;
    int kb2 = (lane >> 4) * 16;   // byte offset of K start
    const char* p = (const char*)&Bs[(colBase + n) * LSTR] + kb2;
    BF16x16 f;
    f.q[0] = *(const uint4*)(p);
    f.q[1] = *(const uint4*)(p + 32);
    return f.v;
}

__device__ __forceinline__ void stage_tiles(unsigned short* As, unsigned short* Bs,
                                            const unsigned short* __restrict__ A,
                                            const unsigned short* __restrict__ B,
                                            int N, int K, int m0, int n0, int k0,
                                            int tid) {
    // A tile 128x32 bf16 = 512 x b128 chunks, 2 per thread (async when available)
    #pragma unroll
    for (int i = 0; i < 2; ++i) {
        int c  = tid + 256 * i;
        int r  = c >> 2;
        int k8 = (c & 3) * 8;
        copy16_g2l(A + (size_t)(m0 + r) * K + (k0 + k8), &As[r * LSTR + k8]);
    }
    // B tile 32x128 bf16 -> transposed into Bs[n][k]
    #pragma unroll
    for (int i = 0; i < 2; ++i) {
        int c  = tid + 256 * i;
        int kr = c >> 4;
        int n8 = (c & 15) * 8;
        uint4 v = *(const uint4*)(B + (size_t)(k0 + kr) * N + (n0 + n8));
        const unsigned short* pv = (const unsigned short*)&v;
        #pragma unroll
        for (int j = 0; j < 8; ++j)
            Bs[(n8 + j) * LSTR + kr] = pv[j];
    }
}

__global__ __launch_bounds__(256)
void gemm_bf16_wmma(const unsigned short* __restrict__ A,
                    const unsigned short* __restrict__ B,
                    const float* __restrict__ bias, void* __restrict__ Cv,
                    int M, int N, int K, int outBf) {
    __shared__ __align__(16) unsigned short As[2][128 * LSTR];
    __shared__ __align__(16) unsigned short Bs[2][128 * LSTR];

    const int tid  = threadIdx.x;
    const int lane = tid & 31;
    const int wave = tid >> 5;
    const int mw   = wave >> 1;          // 0..3  (M within block)
    const int nw   = wave & 1;           // 0..1  (N within block, 64 wide)
    const int m0   = blockIdx.y * 128;
    const int n0   = blockIdx.x * 128;

    v8f acc[2][4] = {};

    const int nk = K >> 5;
    stage_tiles(As[0], Bs[0], A, B, N, K, m0, n0, 0, tid);

    int buf = 0;
    for (int kt = 0; kt < nk; ++kt) {
#if USE_ASYNC_LDS
        __builtin_amdgcn_s_wait_asynccnt(0);
#endif
        __syncthreads();
        if (kt + 1 < nk)
            stage_tiles(As[buf ^ 1], Bs[buf ^ 1], A, B, N, K,
                        m0, n0, (kt + 1) * 32, tid);

        v16bf a0 = load_a_frag(As[buf], mw * 32,      lane);
        v16bf a1 = load_a_frag(As[buf], mw * 32 + 16, lane);
        #pragma unroll
        for (int j = 0; j < 4; ++j) {
            v16bf b = load_b_frag(Bs[buf], nw * 64 + j * 16, lane);
            acc[0][j] = __builtin_amdgcn_wmma_f32_16x16x32_bf16(
                false, a0, false, b, (short)0, acc[0][j], false, false);
            acc[1][j] = __builtin_amdgcn_wmma_f32_16x16x32_bf16(
                false, a1, false, b, (short)0, acc[1][j], false, false);
        }
        buf ^= 1;
    }

    // epilogue: D layout — VGPR r: M=r (lanes 0-15) / M=r+8 (lanes 16-31)
    const int rowAdd = (lane >> 4) << 3;
    const int colIn  = lane & 15;
    float*          Cf = (float*)Cv;
    unsigned short* Cb = (unsigned short*)Cv;
    #pragma unroll
    for (int ti = 0; ti < 2; ++ti) {
        #pragma unroll
        for (int tj = 0; tj < 4; ++tj) {
            F8 a; a.v = acc[ti][tj];
            int col = n0 + nw * 64 + tj * 16 + colIn;
            float bv = bias ? bias[col] : 0.f;
            int rb = m0 + mw * 32 + ti * 16 + rowAdd;
            #pragma unroll
            for (int r = 0; r < 8; ++r) {
                float v = a.f[r] + bv;
                size_t idx = (size_t)(rb + r) * N + col;
                if (outBf) Cb[idx] = f2bf(v);
                else       Cf[idx] = v;
            }
        }
    }
}

// --------------------------- element-wise kernels ---------------------------

__global__ __launch_bounds__(256)
void f32_to_bf16_kernel(const float* __restrict__ in,
                        unsigned short* __restrict__ out, int n4) {
    int i = blockIdx.x * 256 + threadIdx.x;
    if (i < n4) {
        float4 v = ((const float4*)in)[i];
        ushort4 o;
        o.x = f2bf(v.x); o.y = f2bf(v.y); o.z = f2bf(v.z); o.w = f2bf(v.w);
        ((ushort4*)out)[i] = o;
    }
}

// LayerNorm + exact GELU, in place on bf16 rows of length N.
__global__ __launch_bounds__(256)
void ln_gelu_bf16_kernel(unsigned short* __restrict__ X,
                         const float* __restrict__ g,
                         const float* __restrict__ be, int N) {
    unsigned short* x = X + (size_t)blockIdx.x * N;
    float s = 0.f, s2 = 0.f;
    for (int i = threadIdx.x; i < N; i += 256) {
        float v = bf2f(x[i]); s += v; s2 += v * v;
    }
    float S  = block_reduce_sum(s);
    float S2 = block_reduce_sum(s2);
    float mean = S / (float)N;
    float var  = S2 / (float)N - mean * mean;
    float inv  = rsqrtf(var + 1e-5f);
    for (int i = threadIdx.x; i < N; i += 256) {
        float y = (bf2f(x[i]) - mean) * inv * g[i] + be[i];
        x[i] = f2bf(0.5f * y * (1.f + erff(y * 0.70710678118654752f)));
    }
}

// z = normalize(mu + eps*exp(0.5*lv)) -> out (M x 256 f32).  ip is M x 512 f32.
__global__ __launch_bounds__(256)
void reparam_kernel(const float* __restrict__ ip, const float* __restrict__ eps,
                    float* __restrict__ zout) {
    size_t row = blockIdx.x;
    int d = threadIdx.x;
    float mu = ip[row * 512 + d];
    float lv = ip[row * 512 + 256 + d];
    float z  = mu + eps[row * 256 + d] * expf(0.5f * lv);
    float n2 = block_reduce_sum(z * z);
    float n  = sqrtf(n2);
    zout[row * 256 + d] = z / fmaxf(n, 1e-12f);
}

__global__ __launch_bounds__(256)
void kl_accum(const float* __restrict__ ip, float* __restrict__ klsum) {
    size_t row = (size_t)blockIdx.x * SDIM + (SDIM - 1);
    int d = threadIdx.x;
    float mu = ip[row * 512 + d];
    float lv = ip[row * 512 + 256 + d];
    float s  = block_reduce_sum(1.f + lv - mu * mu - expf(lv));
    if (threadIdx.x == 0) atomicAdd(klsum, -0.5f * s);
}

// last-token rows of z (M x 256) -> zl f32 and bf16 (1024 x 256).
__global__ __launch_bounds__(256)
void gather_last(const float* __restrict__ z, float* __restrict__ zl,
                 unsigned short* __restrict__ zlbf) {
    size_t src = ((size_t)blockIdx.x * SDIM + (SDIM - 1)) * 256 + threadIdx.x;
    size_t dst = (size_t)blockIdx.x * 256 + threadIdx.x;
    float v = z[src];
    zl[dst]   = v;
    zlbf[dst] = f2bf(v);
}

__global__ __launch_bounds__(256)
void rowdot_kernel(const float* __restrict__ a, const float* __restrict__ b,
                   float* __restrict__ out, int D) {
    size_t r = blockIdx.x;
    float s = 0.f;
    for (int i = threadIdx.x; i < D; i += 256) s += a[r * D + i] * b[r * D + i];
    s = block_reduce_sum(s);
    if (threadIdx.x == 0) out[r] = s;
}

// (R x C f32) -> (C x R bf16)
__global__ __launch_bounds__(256)
void transpose_bf16_kernel(const float* __restrict__ in,
                           unsigned short* __restrict__ out, int R, int C) {
    int gid = blockIdx.x * 256 + threadIdx.x;
    if (gid < R * C) {
        int r = gid / C, c = gid % C;
        out[(size_t)c * R + r] = f2bf(in[gid]);
    }
}

__global__ __launch_bounds__(256)
void d2s_kernel(const float* __restrict__ xy, const float* __restrict__ sumx,
                const float* __restrict__ sumy, float* __restrict__ d2s) {
    int gid = blockIdx.x * 256 + threadIdx.x;   // 512*512 elements
    int i = gid >> 9, j = gid & 511;
    d2s[gid] = sumx[i] + sumy[j] - 2.f * xy[(size_t)i * 1024 + j];
}

__global__ void init_scalars(Scal* sc) {
    sc->klsum[0] = 0.f; sc->klsum[1] = 0.f;
    sc->alignsum = 0.f;
    sc->lo = 0.f; sc->hi = 8.f; sc->cnt = 0.f;   // unit vectors => d^2 <= 4
}

__global__ __launch_bounds__(256)
void count_le(const float* __restrict__ d2s, int n, Scal* sc) {
    float mid = 0.5f * (sc->lo + sc->hi);
    float c = 0.f;
    for (int i = blockIdx.x * 256 + threadIdx.x; i < n; i += 256 * 256)
        c += (d2s[i] <= mid) ? 1.f : 0.f;
    c = block_reduce_sum(c);
    if (threadIdx.x == 0) atomicAdd(&sc->cnt, c);
}

__global__ void bisect_update(Scal* sc, float target) {
    float mid = 0.5f * (sc->lo + sc->hi);
    if (sc->cnt >= target) sc->hi = mid; else sc->lo = mid;
    sc->cnt = 0.f;
}

__global__ void finalize_bw(Scal* sc) {
    float bw2 = 0.5f * (sc->lo + sc->hi);
    float bw  = sqrtf(fmaxf(bw2, 0.f) * 0.5f);   // * BW_FACTOR(=1)
    bw = fminf(fmaxf(bw, 0.1f), 10.f);
    float ib2 = 1.f / (bw * bw);
    sc->inv_bw2 = ib2; sc->inv_2bw2 = 0.5f * ib2; sc->inv_bw4 = ib2 * ib2;
}

__global__ __launch_bounds__(256)
void stein_combine(const float* __restrict__ xy, const float* __restrict__ ss,
                   const float* __restrict__ xsy, const float* __restrict__ sumx,
                   const float* __restrict__ sumy, const float* __restrict__ ysy,
                   Scal* sc) {
    int gid = blockIdx.x * 256 + threadIdx.x;    // 1024*1024 elements
    int i = gid >> 10, j = gid & 1023;
    float ds = sumx[i] + sumy[j] - 2.f * xy[gid];
    float k  = expf(-ds * sc->inv_2bw2);
    float t  = k * ss[gid]
             + (xsy[gid] - ysy[j]) * k * sc->inv_bw2
             + ds * k * sc->inv_bw4 - k * (256.f * sc->inv_bw2);
    t = block_reduce_sum(t);
    if (threadIdx.x == 0) atomicAdd(&sc->alignsum, t);
}

__global__ void finalize_out(const Scal* sc, float* out_scalars) {
    float align = sc->alignsum * (1.f / 1048576.f);
    float kl    = 0.5f * (sc->klsum[0] + sc->klsum[1]) * (1.f / 1024.f);
    out_scalars[0] = align;
    out_scalars[1] = kl;
    out_scalars[2] = kl + 1.0f * align;          // BETA = 1
}

// ---------------------------------------------------------------------------
// Host launch.  d_in order (jax pytree: dict keys sorted alphabetically):
//   0 image_features 1 text_features 2 eps_img 3 eps_txt 4 attention_mask
//   params: img{5:W1 6:W2 7:b1 8:b2 9:be 10:g}
//           iscore{11..16}  tscore{17..22}  txt{23..28}
// ---------------------------------------------------------------------------
extern "C" void kernel_launch(void* const* d_in, const int* in_sizes, int n_in,
                              void* d_out, int out_size, void* d_ws, size_t ws_size,
                              hipStream_t stream) {
    (void)in_sizes; (void)n_in; (void)out_size; (void)ws_size;

    typedef unsigned short u16;
    char* w = (char*)d_ws;
    auto carve = [&](size_t bytes) -> void* {
        void* p = (void*)w;
        w += (bytes + 255) & ~(size_t)255;
        return p;
    };
    // Xbf (96 MB, live through GEMM1) and ip (128 MB, born at GEMM2) alias.
    void*  shared_reg = carve((size_t)MROWS * 512 * 4);      // 128 MB
    u16*   Xbf   = (u16*)shared_reg;                         // MROWS x 768 bf16
    float* ipb   = (float*)shared_reg;                       // MROWS x 512 f32
    u16*   H1bf  = (u16*)carve((size_t)MROWS * HDIM * 2);    // 128 MB
    u16*   eW1bf = (u16*)carve((size_t)PDIM * HDIM * 2);
    u16*   eW2bf = (u16*)carve((size_t)HDIM * 512 * 2);
    u16*   sW1bf = (u16*)carve(256 * 512 * 2);
    u16*   sW2bf = (u16*)carve(512 * 256 * 2);
    float* zl0   = (float*)carve(1024 * 256 * 4);
    float* zl1   = (float*)carve(1024 * 256 * 4);
    u16*   zl0bf = (u16*)carve(1024 * 256 * 2);
    u16*   zl1bf = (u16*)carve(1024 * 256 * 2);
    float* scl0  = (float*)carve(1024 * 256 * 4);
    float* scl1  = (float*)carve(1024 * 256 * 4);
    u16*   scl0bf= (u16*)carve(1024 * 256 * 2);
    u16*   s1bf  = (u16*)carve(1024 * 512 * 2);
    u16*   tzTbf = (u16*)carve(256 * 1024 * 2);
    u16*   tsTbf = (u16*)carve(256 * 1024 * 2);
    float* xyT   = (float*)carve(1024 * 1024 * 4);
    float* ssT   = (float*)carve(1024 * 1024 * 4);
    float* xsyT  = (float*)carve(1024 * 1024 * 4);
    float* d2s   = (float*)carve(512 * 512 * 4);
    float* sumx  = (float*)carve(1024 * 4);
    float* sumy  = (float*)carve(1024 * 4);
    float* ysy   = (float*)carve(1024 * 4);
    Scal*  sc    = (Scal*)carve(sizeof(Scal));

    auto gemm = [&](const u16* A, const u16* B, const float* bias, void* C,
                    int M, int N, int K, int outBf) {
        dim3 grid(N / 128, M / 128);
        gemm_bf16_wmma<<<grid, 256, 0, stream>>>(A, B, bias, C, M, N, K, outBf);
    };
    auto cvt = [&](const float* in, u16* out, size_t n) {
        int n4 = (int)(n / 4);
        f32_to_bf16_kernel<<<(n4 + 255) / 256, 256, 0, stream>>>(in, out, n4);
    };

    init_scalars<<<1, 1, 0, stream>>>(sc);

    float* zl[2]   = {zl0, zl1};
    u16*   zlbf[2] = {zl0bf, zl1bf};
    float* scl[2]  = {scl0, scl1};

    for (int mod = 0; mod < 2; ++mod) {
        const float* X   = (const float*)d_in[mod];
        const float* eps = (const float*)d_in[2 + mod];
        const int pe = (mod == 0) ? 5 : 23;
        const int ps = (mod == 0) ? 11 : 17;
        const float* eW1 = (const float*)d_in[pe + 0];
        const float* eW2 = (const float*)d_in[pe + 1];
        const float* eb1 = (const float*)d_in[pe + 2];
        const float* eb2 = (const float*)d_in[pe + 3];
        const float* ebe = (const float*)d_in[pe + 4];
        const float* eg  = (const float*)d_in[pe + 5];
        const float* sW1 = (const float*)d_in[ps + 0];
        const float* sW2 = (const float*)d_in[ps + 1];
        const float* sb1 = (const float*)d_in[ps + 2];
        const float* sb2 = (const float*)d_in[ps + 3];
        const float* sbe = (const float*)d_in[ps + 4];
        const float* sg  = (const float*)d_in[ps + 5];

        float* z_out = (float*)d_out + (size_t)mod * BSD;   // iz / tz

        // bf16 operand copies
        cvt(X,   Xbf,   (size_t)MROWS * PDIM);
        cvt(eW1, eW1bf, (size_t)PDIM * HDIM);
        cvt(eW2, eW2bf, (size_t)HDIM * 512);
        cvt(sW1, sW1bf, 256 * 512);
        cvt(sW2, sW2bf, 512 * 256);

        // Encoder MLP (H1 kept bf16, LN+GELU in place)
        gemm(Xbf, eW1bf, eb1, H1bf, MROWS, HDIM, PDIM, 1);
        ln_gelu_bf16_kernel<<<MROWS, 256, 0, stream>>>(H1bf, eg, ebe, HDIM);
        gemm(H1bf, eW2bf, eb2, ipb, MROWS, 512, HDIM, 0);   // overwrites Xbf (dead)

        // Reparameterize + normalize -> output, KL on last rows
        reparam_kernel<<<MROWS, 256, 0, stream>>>(ipb, eps, z_out);
        kl_accum<<<BDIM, 256, 0, stream>>>(ipb, &sc->klsum[mod]);

        // Score MLP only on last-token rows
        gather_last<<<BDIM, 256, 0, stream>>>(z_out, zl[mod], zlbf[mod]);
        gemm(zlbf[mod], sW1bf, sb1, s1bf, 1024, 512, 256, 1);
        ln_gelu_bf16_kernel<<<1024, 256, 0, stream>>>(s1bf, sg, sbe, 512);
        gemm(s1bf, sW2bf, sb2, scl[mod], 1024, 256, 512, 0);
    }

    // Stein discrepancy pieces
    cvt(scl0, scl0bf, 1024 * 256);
    transpose_bf16_kernel<<<(1024 * 256) / 256, 256, 0, stream>>>(zl1, tzTbf, 1024, 256);
    transpose_bf16_kernel<<<(1024 * 256) / 256, 256, 0, stream>>>(scl1, tsTbf, 1024, 256);
    rowdot_kernel<<<1024, 256, 0, stream>>>(zl0, zl0, sumx, 256);
    rowdot_kernel<<<1024, 256, 0, stream>>>(zl1, zl1, sumy, 256);
    rowdot_kernel<<<1024, 256, 0, stream>>>(zl1, scl1, ysy, 256);

    gemm(zl0bf,  tzTbf, nullptr, xyT,  1024, 1024, 256, 0);   // x @ y.T
    gemm(scl0bf, tsTbf, nullptr, ssT,  1024, 1024, 256, 0);   // sx @ sy.T
    gemm(zl0bf,  tsTbf, nullptr, xsyT, 1024, 1024, 256, 0);   // x @ sy.T

    // Bandwidth: median of 512x512 sample distances via device bisection
    d2s_kernel<<<(512 * 512) / 256, 256, 0, stream>>>(xyT, sumx, sumy, d2s);
    const float target = 0.5f * 512.f * 512.f;
    for (int it = 0; it < 30; ++it) {
        count_le<<<256, 256, 0, stream>>>(d2s, 512 * 512, sc);
        bisect_update<<<1, 1, 0, stream>>>(sc, target);
    }
    finalize_bw<<<1, 1, 0, stream>>>(sc);

    stein_combine<<<(1024 * 1024) / 256, 256, 0, stream>>>(xyT, ssT, xsyT,
                                                           sumx, sumy, ysy, sc);
    finalize_out<<<1, 1, 0, stream>>>(sc, (float*)d_out + 2 * BSD);
}
